// Z2EquivariantRNN_83305185673578
// MI455X (gfx1250) — compile-verified, matrix operand order checked by reading, not run on hardware
//
#include <hip/hip_runtime.h>
#include <hip/hip_bf16.h>

// Problem constants (match reference)
#define TSEQ 1024
#define NB   256
#define HD   64
#define ID   128

typedef __attribute__((ext_vector_type(16))) __bf16 v16bf;
typedef __attribute__((ext_vector_type(8)))  float  v8f;

// f32 -> bf16 via hardware convert (v_cvt_*_bf16_f32 on gfx1250, RTNE)
__device__ __forceinline__ __bf16 to_bf16(float f) {
  return (__bf16)f;
}

// hardware tanh if available (V_TANH_F32 is a CDNA5 TRANS32 op)
__device__ __forceinline__ float fast_tanh(float x) {
#if __has_builtin(__builtin_amdgcn_tanhf)
  return __builtin_amdgcn_tanhf(x);
#elif __has_builtin(__builtin_amdgcn_tanh_f32)
  return __builtin_amdgcn_tanh_f32(x);
#else
  return tanhf(x);
#endif
}

// 16-bit A/B matrix VGPR layout (ISA 7.12.2): lane L holds row (A) / col (B)
// = L&15; K offsets: half-wave base 0/8, vgprs 0-3 -> base+{0..7},
// vgprs 4-7 -> base+16+{0..7}; each 32b vgpr = 2 bf16 (K even, K odd).
__device__ __forceinline__ int k_map(int lane, int e) {
  int v = e >> 1, j = e & 1;
  int base = (lane < 16) ? 0 : 8;
  return (v < 4) ? (base + 2 * v + j) : (base + 16 + 2 * (v - 4) + j);
}

__device__ __forceinline__ v8f wmma_bf16(v16bf a, v16bf b, v8f c) {
  // D = A(16x32 bf16) x B(32x16 bf16) + C(16x16 f32)
  return __builtin_amdgcn_wmma_f32_16x16x32_bf16(
      /*neg_a=*/false, a, /*neg_b=*/false, b,
      /*c_mod=*/(short)0, c, /*reuse_a=*/false, /*reuse_b=*/false);
}

// ---------------------------------------------------------------------------
// Kernel 1: W_comm[i][j] = (P+ A+ P+ + P- A- P-)[i][j]; s(k) = k^32.
// ---------------------------------------------------------------------------
__global__ void wcomm_kernel(const float* __restrict__ Ap,
                             const float* __restrict__ Am,
                             float* __restrict__ Wc) {
  int idx = blockIdx.x * blockDim.x + threadIdx.x;
  if (idx >= HD * HD) return;
  int i = idx >> 6, j = idx & 63;
  int si = i ^ 32, sj = j ^ 32;
  float w = 0.25f * (Ap[i * HD + j] + Ap[si * HD + j] + Ap[i * HD + sj] + Ap[si * HD + sj]
                   + Am[i * HD + j] - Am[si * HD + j] - Am[i * HD + sj] + Am[si * HD + sj]);
  Wc[idx] = w;
}

// ---------------------------------------------------------------------------
// Kernel 2: inp[t][b][h] = x[b][t][:] . W_in_w[h][:] + (W_in_b+bias)[h]
// M = B*T rows, K=128 (4 k-tiles), N=64 (4 n-tiles). bf16 WMMA, f32 acc.
// 256 blocks x 8 waves x 8 m-tiles x 16 rows = 262144 rows.
// HBM-bound: streams 128 MiB of x once; converts f32->bf16 in registers.
// ---------------------------------------------------------------------------
__global__ void __launch_bounds__(256) inproj_kernel(
    const float* __restrict__ x, const float* __restrict__ Ww,
    const float* __restrict__ Wb, const float* __restrict__ bias,
    float* __restrict__ inp) {
  __shared__ __bf16 sW[ID * HD];  // [k][h] = W_in_w[h][k], bf16
  __shared__ float sBias[HD];
  int tid = threadIdx.x;
  for (int idx = tid; idx < ID * HD; idx += 256) {
    int k = idx >> 6, h = idx & 63;
    sW[idx] = to_bf16(Ww[h * ID + k]);
  }
  if (tid < HD) sBias[tid] = Wb[tid] + bias[tid];
  __syncthreads();

  int lane = tid & 31, wave = tid >> 5;
  int l15 = lane & 15, hh = lane >> 4;
  int kbase = hh ? 8 : 0;

  // Hoist all 4x4 B tiles of W_in into registers (shared by all m-tiles)
  v16bf wb[4][4];
#pragma unroll
  for (int kt = 0; kt < 4; ++kt)
#pragma unroll
    for (int nt = 0; nt < 4; ++nt)
#pragma unroll
      for (int e = 0; e < 16; ++e)
        wb[kt][nt][e] = sW[(kt * 32 + k_map(lane, e)) * HD + nt * 16 + l15];

  for (int it = 0; it < 8; ++it) {
    int mtile = (blockIdx.x * 8 + wave) * 8 + it;
    long row = (long)mtile * 16 + l15;

    // Prefetch next iteration's A rows into cache (global_prefetch_b8)
    if (it < 7) {
      const float* nxt = x + (row + 16) * ID;
      __builtin_prefetch(nxt, 0, 3);
      __builtin_prefetch(nxt + 64, 0, 3);
    }

    v8f acc[4] = {};
#pragma unroll
    for (int kt = 0; kt < 4; ++kt) {
      // A tile: 16 rows x 32 cols of x, converted f32->bf16 in registers
      v16bf a;
      const float* xr = x + row * ID + kt * 32;
#pragma unroll
      for (int v = 0; v < 8; ++v) {
        int kk = (v < 4) ? (kbase + 2 * v) : (kbase + 16 + 2 * (v - 4));
        float2 p = *reinterpret_cast<const float2*>(xr + kk);  // global_load_b64
        a[2 * v]     = to_bf16(p.x);
        a[2 * v + 1] = to_bf16(p.y);
      }
#pragma unroll
      for (int nt = 0; nt < 4; ++nt)
        acc[nt] = wmma_bf16(a, wb[kt][nt], acc[nt]);
    }
    // Epilogue: add bias, scatter to (T,B,H) layout for the scan kernel
#pragma unroll
    for (int nt = 0; nt < 4; ++nt) {
      int n = nt * 16 + l15;
#pragma unroll
      for (int i = 0; i < 8; ++i) {
        int m = i + hh * 8;
        long r = (long)mtile * 16 + m;       // flat (b,t) row of x
        int b = (int)(r >> 10);              // r / TSEQ
        int t = (int)(r & 1023);             // r % TSEQ
        inp[((long)t * NB + b) * HD + n] = acc[nt][i] + sBias[n];
      }
    }
  }
}

// ---------------------------------------------------------------------------
// Kernel 3: sequential scan h = tanh(h @ W_comm^T + u_t). Batch rows are
// independent -> 16 single-wave workgroups, each owns 16 batch rows and keeps
// W_comm (bf16 B-layout) in registers. h is converted C-layout(f32) ->
// A-layout(bf16) each step via a 4KB LDS tile.
// ---------------------------------------------------------------------------
__global__ void __launch_bounds__(32) recurrence_kernel(
    const float* __restrict__ inp, const float* __restrict__ Wc,
    float* __restrict__ hidden_out) {
  __shared__ float sh[16 * HD];
  int lane = threadIdx.x & 31;
  int l15 = lane & 15, hh = lane >> 4;
  int b0 = blockIdx.x * 16;

  // B[k][n] = W_comm[n][k]  (h @ W_comm^T)
  v16bf wc[2][4];
#pragma unroll
  for (int kt = 0; kt < 2; ++kt)
#pragma unroll
    for (int nt = 0; nt < 4; ++nt)
#pragma unroll
      for (int e = 0; e < 16; ++e) {
        int n = nt * 16 + l15;
        int k = kt * 32 + k_map(lane, e);
        wc[kt][nt][e] = to_bf16(Wc[n * HD + k]);
      }

  v16bf a[2] = {};      // h0 = 0 in A-layout
  v8f hstate[4] = {};
  for (int t = 0; t < TSEQ; ++t) {
    v8f acc[4];
#pragma unroll
    for (int nt = 0; nt < 4; ++nt) {
      int n = nt * 16 + l15;
#pragma unroll
      for (int i = 0; i < 8; ++i) {
        int m = i + hh * 8;
        acc[nt][i] = inp[((long)t * NB + (b0 + m)) * HD + n];  // u_t
      }
    }
#pragma unroll
    for (int kt = 0; kt < 2; ++kt)
#pragma unroll
      for (int nt = 0; nt < 4; ++nt)
        acc[nt] = wmma_bf16(a[kt], wc[kt][nt], acc[nt]);
    // tanh (hardware V_TANH_F32), keep f32 state, stage into LDS for
    // C-layout -> A-layout transpose
#pragma unroll
    for (int nt = 0; nt < 4; ++nt) {
      int n = nt * 16 + l15;
#pragma unroll
      for (int i = 0; i < 8; ++i) {
        int m = i + hh * 8;
        float hv = fast_tanh(acc[nt][i]);
        hstate[nt][i] = hv;
        sh[m * HD + n] = hv;
      }
    }
    __syncthreads();  // single-wave WG: S_NOP; DS in-order makes this safe
#pragma unroll
    for (int kt = 0; kt < 2; ++kt)
#pragma unroll
      for (int e = 0; e < 16; ++e)
        a[kt][e] = to_bf16(sh[l15 * HD + kt * 32 + k_map(lane, e)]);
    __syncthreads();
  }
  // final hidden state
#pragma unroll
  for (int nt = 0; nt < 4; ++nt) {
    int n = nt * 16 + l15;
#pragma unroll
    for (int i = 0; i < 8; ++i) {
      int m = i + hh * 8;
      hidden_out[(long)(b0 + m) * HD + n] = hstate[nt][i];
    }
  }
}

// ---------------------------------------------------------------------------
// Kernel 4: predictions = hidden @ W_out_w^T + W_out_b. (256x64)x(64x128).
// One block, 16 waves, one 16-row m-tile each; 8 n-tiles x 2 k-tiles WMMA.
// ---------------------------------------------------------------------------
__global__ void __launch_bounds__(512) outproj_kernel(
    const float* __restrict__ hidden, const float* __restrict__ Ww,
    const float* __restrict__ Wb, float* __restrict__ pred) {
  __shared__ __bf16 sW[HD * ID];  // [k][n] = W_out_w[n][k]
  __shared__ float sBias[ID];
  int tid = threadIdx.x;
  for (int idx = tid; idx < HD * ID; idx += 512) {
    int k = idx >> 7, n = idx & 127;
    sW[idx] = to_bf16(Ww[n * HD + k]);
  }
  if (tid < ID) sBias[tid] = Wb[tid];
  __syncthreads();

  int lane = tid & 31, wave = tid >> 5;
  int l15 = lane & 15, hh = lane >> 4;
  int mtile = wave;  // 16 waves x 16 rows = 256 batch rows

  v16bf a[2];
#pragma unroll
  for (int kt = 0; kt < 2; ++kt)
#pragma unroll
    for (int e = 0; e < 16; ++e)
      a[kt][e] = to_bf16(hidden[(long)(mtile * 16 + l15) * HD + kt * 32 + k_map(lane, e)]);

  v8f acc[8] = {};
#pragma unroll
  for (int kt = 0; kt < 2; ++kt)
#pragma unroll
    for (int nt = 0; nt < 8; ++nt) {
      v16bf bm;
#pragma unroll
      for (int e = 0; e < 16; ++e)
        bm[e] = sW[(kt * 32 + k_map(lane, e)) * ID + nt * 16 + l15];
      acc[nt] = wmma_bf16(a[kt], bm, acc[nt]);
    }
#pragma unroll
  for (int nt = 0; nt < 8; ++nt) {
    int n = nt * 16 + l15;
#pragma unroll
    for (int i = 0; i < 8; ++i) {
      int m = i + hh * 8;
      pred[(long)(mtile * 16 + m) * ID + n] = acc[nt][i] + sBias[n];
    }
  }
}

// ---------------------------------------------------------------------------
extern "C" void kernel_launch(void* const* d_in, const int* in_sizes, int n_in,
                              void* d_out, int out_size, void* d_ws, size_t ws_size,
                              hipStream_t stream) {
  const float* x    = (const float*)d_in[0];
  const float* Ap   = (const float*)d_in[1];
  const float* Am   = (const float*)d_in[2];
  const float* Wiw  = (const float*)d_in[3];
  const float* Wib  = (const float*)d_in[4];
  const float* Wow  = (const float*)d_in[5];
  const float* Wob  = (const float*)d_in[6];
  const float* bias = (const float*)d_in[7];

  float* out    = (float*)d_out;
  float* pred   = out;                 // (256,128)
  float* hidden = out + NB * ID;       // (256,64)

  // Workspace: inp (T,B,H) f32 = 64 MiB, then W_comm (64x64) f32.
  float* inp = (float*)d_ws;
  float* Wc  = inp + (size_t)TSEQ * NB * HD;

  wcomm_kernel<<<16, 256, 0, stream>>>(Ap, Am, Wc);
  inproj_kernel<<<256, 256, 0, stream>>>(x, Wiw, Wib, bias, inp);
  recurrence_kernel<<<16, 32, 0, stream>>>(inp, Wc, hidden);
  outproj_kernel<<<1, 512, 0, stream>>>(hidden, Wow, Wob, pred);
}
